// LSTM_45165876085602
// MI455X (gfx1250) — matrix operand order repaired and checked by reading, not compiled, not run on hardware
//
#include <hip/hip_runtime.h>
#include <stdint.h>

// ---------------------------------------------------------------------------
// LSTM (B=32, T=2048, D=H=512, L=2) for gfx1250 via bf16 WMMA.
//   Phase A: Xp = X @ Wih^T + (b_ih+b_hh)  — 32x64 macro-tile WMMA GEMM.
//   Phase B: ONE persistent kernel per layer: 16 WGs x 4 waves, Whh LDS-resident,
//            device-scope atomic barrier between timesteps.
// ---------------------------------------------------------------------------

#define B_  32
#define T_  2048
#define D_  512
#define H_  512
#define L_  2
#define G4H 2048     // 4*H
#define NWG 16       // workgroups in persistent recurrence kernel
#define LPITCH 520   // LDS row pitch (elements): 512+8 -> ds_load_b128 hits all 64 banks

typedef __attribute__((ext_vector_type(16))) __bf16 v16bf;
typedef __attribute__((ext_vector_type(8)))  float  v8f;

__device__ __forceinline__ unsigned short f2bf(float f) {
    unsigned int u = __float_as_uint(f);
    unsigned int r = u + 0x7FFFu + ((u >> 16) & 1u);   // round-to-nearest-even
    return (unsigned short)(r >> 16);
}

__device__ __forceinline__ float sigf(float x) {
    // sigmoid via v_tanh_f32 (gfx1250 transcendental)
    return 0.5f * tanhf(0.5f * x) + 0.5f;
}

// ---------------------------------------------------------------------------
// Utility kernels
// ---------------------------------------------------------------------------
__global__ void k_cvt_bf16(const float* __restrict__ s, unsigned short* __restrict__ d, int n) {
    int i = blockIdx.x * blockDim.x + threadIdx.x;
    int stride = gridDim.x * blockDim.x;
    for (; i < n; i += stride) d[i] = f2bf(s[i]);
}

__global__ void k_fold_bias(const float* __restrict__ bih, const float* __restrict__ bhh,
                            float* __restrict__ b, int n) {
    int i = blockIdx.x * blockDim.x + threadIdx.x;
    int stride = gridDim.x * blockDim.x;
    for (; i < n; i += stride) b[i] = bih[i] + bhh[i];
}

__global__ void k_init_state(const float* __restrict__ h0, const float* __restrict__ c0,
                             unsigned short* __restrict__ hbf, float* __restrict__ c, int n) {
    int i = blockIdx.x * blockDim.x + threadIdx.x;
    int stride = gridDim.x * blockDim.x;
    for (; i < n; i += stride) { hbf[i] = f2bf(h0[i]); c[i] = c0[i]; }
}

__global__ void k_zero_bar(unsigned* __restrict__ bar) { bar[0] = 0u; bar[1] = 0u; }

// ---------------------------------------------------------------------------
// Phase A: Xp[r, n] = sum_k A(r,k) * Wih[n,k] + bias[n]
//   rows r = t*B + b (65536), cols n (2048), K = 512.
//   permA=1: A(r,k) = x_bf[(b*T + t)*D + k]   (layer 0, x kept [B,T,D])
//   permA=0: A(r,k) = seq_bf[r*H + k]         (layer 1)
// One wave per 32x64 macro-tile (2 M-tiles x 4 N-tiles): A reused x4, B reused x2.
// ---------------------------------------------------------------------------
__global__ void __launch_bounds__(256)
k_gemm_xp(const unsigned short* __restrict__ A,
          const unsigned short* __restrict__ W,     // [4H, 512] bf16 (row n, col k)
          const float* __restrict__ bias,           // [4H]
          float* __restrict__ Xp,                   // [65536, 2048] f32
          int permA)
{
    const int lane = threadIdx.x & 31;
    const int wave = threadIdx.x >> 5;
    const int tile = blockIdx.x * 8 + wave;         // 65536 macro-tiles
    const int tn4  = tile & 31;                     // 64-col block (2048/64)
    const int tm2  = tile >> 5;                     // 32-row block (65536/32)
    const int g    = lane >> 4;
    const int l15  = lane & 15;

    const int r0 = tm2 * 32 + l15;                  // A rows for the two M-tiles
    const int r1 = r0 + 16;
    size_t ab0, ab1;
    if (permA) {
        ab0 = ((size_t)(r0 & 31) * T_ + (r0 >> 5)) * (size_t)D_;
        ab1 = ((size_t)(r1 & 31) * T_ + (r1 >> 5)) * (size_t)D_;
    } else {
        ab0 = (size_t)r0 * H_;
        ab1 = (size_t)r1 * H_;
    }
    size_t bb[4];
#pragma unroll
    for (int s = 0; s < 4; ++s) bb[s] = (size_t)(tn4 * 64 + s * 16 + l15) * 512;

    v8f acc0[4] = {}, acc1[4] = {};
    for (int k0 = 0; k0 < 512; k0 += 32) {
        v16bf a0, a1;
        ((uint4*)&a0)[0] = *(const uint4*)(A + ab0 + k0 + g * 16);
        ((uint4*)&a0)[1] = *(const uint4*)(A + ab0 + k0 + g * 16 + 8);
        ((uint4*)&a1)[0] = *(const uint4*)(A + ab1 + k0 + g * 16);
        ((uint4*)&a1)[1] = *(const uint4*)(A + ab1 + k0 + g * 16 + 8);
#pragma unroll
        for (int s = 0; s < 4; ++s) {
            v16bf b;
            const uint4* p = (const uint4*)(W + bb[s] + k0 + g * 16);
            ((uint4*)&b)[0] = p[0];
            ((uint4*)&b)[1] = p[1];
            acc0[s] = __builtin_amdgcn_wmma_f32_16x16x32_bf16(
                          false, a0, false, b, (short)0, acc0[s], false, false);
            acc1[s] = __builtin_amdgcn_wmma_f32_16x16x32_bf16(
                          false, a1, false, b, (short)0, acc1[s], false, false);
        }
    }

#pragma unroll
    for (int s = 0; s < 4; ++s) {
        const int n = tn4 * 64 + s * 16 + l15;      // C col for this lane
        const float bv = bias[n];
#pragma unroll
        for (int vv = 0; vv < 8; ++vv) {
            const int row = tm2 * 32 + vv + 8 * g;  // C layout: M = vv (+8 for hi half)
            Xp[(size_t)row * G4H + n]        = acc0[s][vv] + bv;
            Xp[(size_t)(row + 16) * G4H + n] = acc1[s][vv] + bv;
        }
    }
}

// ---------------------------------------------------------------------------
// Phase B: persistent recurrence kernel. 16 WGs x 128 threads (4 waves).
// Global tile v = wg*4+wave; tn = v>>1 (H col tile 0..31), tm = v&1 (B rows).
// Each WG stages its Whh slab (4 gates x 32 cols x 512 K bf16 = 128 KB) into
// LDS once, then iterates t=0..T-1 with a device-scope barrier per step.
// ---------------------------------------------------------------------------
__global__ void __launch_bounds__(128)
k_lstm_layer(const float* __restrict__ Xp,            // [T*B, 4H]
             const unsigned short* __restrict__ Whh,  // [4H, 512] bf16
             unsigned short* __restrict__ hbf0,       // [B, H] bf16 ping
             unsigned short* __restrict__ hbf1,       // [B, H] bf16 pong
             float* __restrict__ c,                   // [B, H] f32 (persistent)
             unsigned short* __restrict__ seq_bf,     // layer0: [T*B, H] bf16 (or null)
             float* __restrict__ out_seq,             // layer1: [B, T, H] f32 (or null)
             float* __restrict__ hT, float* __restrict__ cT,
             unsigned* __restrict__ bar)              // bar[0]=count, bar[1]=phase
{
    __shared__ unsigned short Bs[128 * LPITCH];       // 133,120 B (one WG per WGP)

    const int wg   = blockIdx.x;                      // 0..15
    const int lane = threadIdx.x & 31;
    const int wave = threadIdx.x >> 5;                // 0..3
    const int v    = wg * 4 + wave;                   // global tile 0..63
    const int tn   = v >> 1;                          // H col tile
    const int tm   = v & 1;                           // B row tile
    const int g    = lane >> 4;
    const int l15  = lane & 15;
    const int jloc = ((v >> 1) & 1) * 16 + l15;       // col within WG's 32-col slab

    // ---- stage Whh slab: LDS row lr = G*32 + j  <->  global row G*512 + wg*32 + j
    for (int idx = threadIdx.x; idx < 128 * 64; idx += 128) {
        const int lr = idx >> 6;
        const int ch = idx & 63;                      // 16B chunk within a 512-elem row
        const int G  = lr >> 5;
        const int j  = lr & 31;
        *(uint4*)&Bs[lr * LPITCH + ch * 8] =
            *(const uint4*)(Whh + (size_t)(G * H_ + wg * 32 + j) * 512 + ch * 8);
    }
    __syncthreads();

    const size_t abase = (size_t)(tm * 16 + l15) * H_;
    const int    n0    = tn * 16;

    for (int t = 0; t < T_; ++t) {
        const unsigned short* hc = (t & 1) ? hbf1 : hbf0;
        unsigned short*       hn = (t & 1) ? hbf0 : hbf1;

        v8f ai = {}, af = {}, ag = {}, ao = {};
        for (int k0 = 0; k0 < 512; k0 += 32) {
            v16bf a;
            ((uint4*)&a)[0] = *(const uint4*)(hc + abase + k0 + g * 16);
            ((uint4*)&a)[1] = *(const uint4*)(hc + abase + k0 + g * 16 + 8);

            const int kb = k0 + g * 16;
            v16bf b0, b1, b2, b3;
            ((uint4*)&b0)[0] = *(const uint4*)&Bs[(0 * 32 + jloc) * LPITCH + kb];
            ((uint4*)&b0)[1] = *(const uint4*)&Bs[(0 * 32 + jloc) * LPITCH + kb + 8];
            ((uint4*)&b1)[0] = *(const uint4*)&Bs[(1 * 32 + jloc) * LPITCH + kb];
            ((uint4*)&b1)[1] = *(const uint4*)&Bs[(1 * 32 + jloc) * LPITCH + kb + 8];
            ((uint4*)&b2)[0] = *(const uint4*)&Bs[(2 * 32 + jloc) * LPITCH + kb];
            ((uint4*)&b2)[1] = *(const uint4*)&Bs[(2 * 32 + jloc) * LPITCH + kb + 8];
            ((uint4*)&b3)[0] = *(const uint4*)&Bs[(3 * 32 + jloc) * LPITCH + kb];
            ((uint4*)&b3)[1] = *(const uint4*)&Bs[(3 * 32 + jloc) * LPITCH + kb + 8];

            ai = __builtin_amdgcn_wmma_f32_16x16x32_bf16(false, a, false, b0, (short)0, ai, false, false);
            af = __builtin_amdgcn_wmma_f32_16x16x32_bf16(false, a, false, b1, (short)0, af, false, false);
            ag = __builtin_amdgcn_wmma_f32_16x16x32_bf16(false, a, false, b2, (short)0, ag, false, false);
            ao = __builtin_amdgcn_wmma_f32_16x16x32_bf16(false, a, false, b3, (short)0, ao, false, false);
        }

#pragma unroll
        for (int vv = 0; vv < 8; ++vv) {
            const int brow = tm * 16 + vv + 8 * g;    // batch row
            const int col  = n0 + l15;                // h column
            const float* xr = Xp + ((size_t)t * B_ + brow) * (size_t)G4H + col;
            float i_ = sigf (ai[vv] + xr[0]);
            float f_ = sigf (af[vv] + xr[512]);
            float g_ = tanhf(ag[vv] + xr[1024]);
            float o_ = sigf (ao[vv] + xr[1536]);
            const size_t ci = (size_t)brow * H_ + col;
            float cn = f_ * c[ci] + i_ * g_;
            c[ci] = cn;
            float h = o_ * tanhf(cn);
            hn[ci] = f2bf(h);
            if (seq_bf) seq_bf[((size_t)t * B_ + brow) * (size_t)H_ + col] = f2bf(h);
            if (out_seq) out_seq[((size_t)brow * T_ + t) * (size_t)H_ + col] = h;
            if (t == T_ - 1) { hT[ci] = h; cT[ci] = cn; }
        }

        // ---- device-scope sense-reversing barrier across the 16 WGs
        __threadfence();            // release this step's h/c writes
        __syncthreads();
        if (threadIdx.x == 0) {
            unsigned cur  = __hip_atomic_load(&bar[1], __ATOMIC_RELAXED, __HIP_MEMORY_SCOPE_AGENT);
            unsigned prev = __hip_atomic_fetch_add(&bar[0], 1u, __ATOMIC_ACQ_REL, __HIP_MEMORY_SCOPE_AGENT);
            if (prev == NWG - 1) {
                __hip_atomic_store(&bar[0], 0u, __ATOMIC_RELAXED, __HIP_MEMORY_SCOPE_AGENT);
                __hip_atomic_fetch_add(&bar[1], 1u, __ATOMIC_RELEASE, __HIP_MEMORY_SCOPE_AGENT);
            } else {
                while (__hip_atomic_load(&bar[1], __ATOMIC_ACQUIRE, __HIP_MEMORY_SCOPE_AGENT) == cur)
                    __builtin_amdgcn_s_sleep(2);
            }
        }
        __syncthreads();
        __threadfence();            // acquire other WGs' h writes
    }
}

// ---------------------------------------------------------------------------
// Workspace layout (bytes)
// ---------------------------------------------------------------------------
static constexpr size_t O_XBF  = 0;                                  // x bf16 [B,T,D]      64 MiB
static constexpr size_t O_SEQ  = O_XBF  + (size_t)B_*T_*D_*2;        // seq bf16 [T*B,H]    64 MiB
static constexpr size_t O_WIH  = O_SEQ  + (size_t)T_*B_*H_*2;        // Wih bf16             4 MiB
static constexpr size_t O_WHH  = O_WIH  + (size_t)L_*G4H*D_*2;       // Whh bf16             4 MiB
static constexpr size_t O_BIAS = O_WHH  + (size_t)L_*G4H*H_*2;       // bias f32            16 KiB
static constexpr size_t O_HBF0 = O_BIAS + (size_t)L_*G4H*4;          // h ping bf16         32 KiB
static constexpr size_t O_HBF1 = O_HBF0 + (size_t)B_*H_*2;           // h pong bf16         32 KiB
static constexpr size_t O_C    = O_HBF1 + (size_t)B_*H_*2;           // c f32               64 KiB
static constexpr size_t O_XP   = O_C    + (size_t)B_*H_*4;           // Xp f32 [T*B,4H]    512 MiB
static constexpr size_t O_BAR  = O_XP   + (size_t)T_*B_*G4H*4;       // barrier: 2x u32

extern "C" void kernel_launch(void* const* d_in, const int* in_sizes, int n_in,
                              void* d_out, int out_size, void* d_ws, size_t ws_size,
                              hipStream_t stream) {
    (void)in_sizes; (void)n_in; (void)out_size; (void)ws_size;

    const float* x    = (const float*)d_in[0];   // [B,T,D]
    const float* h0   = (const float*)d_in[1];   // [L,B,H]
    const float* c0   = (const float*)d_in[2];   // [L,B,H]
    const float* Wih  = (const float*)d_in[3];   // [L,4H,D]
    const float* Whh  = (const float*)d_in[4];   // [L,4H,H]
    const float* bih  = (const float*)d_in[5];   // [L,4H]
    const float* bhh  = (const float*)d_in[6];   // [L,4H]

    uint8_t* ws = (uint8_t*)d_ws;
    unsigned short* x_bf   = (unsigned short*)(ws + O_XBF);
    unsigned short* seq_bf = (unsigned short*)(ws + O_SEQ);
    unsigned short* wih_bf = (unsigned short*)(ws + O_WIH);
    unsigned short* whh_bf = (unsigned short*)(ws + O_WHH);
    float*          biasf  = (float*)(ws + O_BIAS);
    unsigned short* hbf0   = (unsigned short*)(ws + O_HBF0);
    unsigned short* hbf1   = (unsigned short*)(ws + O_HBF1);
    float*          cbuf   = (float*)(ws + O_C);
    float*          xp     = (float*)(ws + O_XP);
    unsigned*       bar    = (unsigned*)(ws + O_BAR);

    float* out_seq = (float*)d_out;                        // [B,T,H]
    float* out_h   = out_seq + (size_t)B_ * T_ * H_;       // [L,B,H]
    float* out_c   = out_h   + (size_t)L_ * B_ * H_;       // [L,B,H]

    // Convert inputs to bf16, fold biases.
    k_cvt_bf16 <<<4096, 256, 0, stream>>>(x,   x_bf,   B_ * T_ * D_);
    k_cvt_bf16 <<<512,  256, 0, stream>>>(Wih, wih_bf, L_ * G4H * D_);
    k_cvt_bf16 <<<512,  256, 0, stream>>>(Whh, whh_bf, L_ * G4H * H_);
    k_fold_bias<<<16,   256, 0, stream>>>(bih, bhh, biasf, L_ * G4H);

    for (int l = 0; l < L_; ++l) {
        k_init_state<<<64, 256, 0, stream>>>(h0 + (size_t)l * B_ * H_,
                                             c0 + (size_t)l * B_ * H_,
                                             hbf0, cbuf, B_ * H_);
        k_zero_bar<<<1, 1, 0, stream>>>(bar);

        // Phase A: all-timestep input projection (65536 macro-tiles / 8 per block).
        k_gemm_xp<<<8192, 256, 0, stream>>>(
            (l == 0) ? x_bf : seq_bf,
            wih_bf + (size_t)l * G4H * D_,
            biasf  + (size_t)l * G4H,
            xp,
            (l == 0) ? 1 : 0);

        // Phase B: whole-layer recurrence in ONE persistent kernel.
        k_lstm_layer<<<NWG, 128, 0, stream>>>(
            xp,
            whh_bf + (size_t)l * G4H * H_,
            hbf0, hbf1, cbuf,
            (l == 0) ? seq_bf : (unsigned short*)nullptr,
            (l == L_ - 1) ? out_seq : (float*)nullptr,
            out_h + (size_t)l * B_ * H_,
            out_c + (size_t)l * B_ * H_,
            bar);
    }
}
